// LSTMDecoder_8572754723489
// MI455X (gfx1250) — matrix operand order repaired and checked by reading, not compiled, not run on hardware
//
#include <hip/hip_runtime.h>
#include <hip/hip_bf16.h>
#include <stdint.h>

// ---------------------------------------------------------------------------
// LSTM decoder for MI455X (gfx1250, wave32, WMMA).
//   L=256 steps, B=256 batch, H=512 hidden, A=I=512.
//   gates = [feat,x_t] @ W_ih^T + b_ih + h @ W_hh^T + b_hh
// Strategy:
//   (1) base[b][n]  = feat @ W_ih[:, :512]^T + b_ih + b_hh            (once)
//   (2) xg[t][b][n] = base + x_t @ W_ih[:, 512:]^T                    (one big GEMM)
//   (3) ONE persistent kernel runs all 256 recurrent steps:
//         32 blocks x 512 threads; block nt owns 16 j'-columns of all 4 gates,
//         W_hh slice (64x512 bf16 = 64KB) lives in LDS, c lives in registers,
//         h double-buffered in global with a grid-wide release/acquire barrier.
// All GEMMs: v_wmma_f32_16x16x32_bf16 (bf16 operands, f32 accumulate).
// GEMM dims are compile-time (template) so every fragment load folds into
// base-pointer + 24-bit immediate offsets (no per-iteration v_add_nc_u64,
// no WMMA->VALU hazard NOPs in the hot loop).
// ---------------------------------------------------------------------------

typedef __attribute__((ext_vector_type(16))) __bf16 v16bf;
typedef __attribute__((ext_vector_type(8)))  __bf16 v8bf;
typedef __attribute__((ext_vector_type(8)))  float  v8f;

#define LSEQ 256
#define BB   256
#define HH   512
#define AA   512
#define II   512
#define G4   2048   // 4*H

__device__ __forceinline__ unsigned short f2bf(float f) {
  union { float f; uint32_t u; } v; v.f = f;
  uint32_t r = v.u + 0x7FFFu + ((v.u >> 16) & 1u);   // round to nearest even
  return (unsigned short)(r >> 16);
}

__device__ __forceinline__ float sigmoidf_(float x) {
  return 1.0f / (1.0f + __expf(-x));
}

// 16-bit A-matrix 16x32 layout (contiguous, matches B): lane ln (row m = ln&15, ksel = ln>>4)
// holds K = 16*ksel..16*ksel+15 in v[0..15].
// `ap` must already include the +16*ksel element offset.
__device__ __forceinline__ v16bf load_a_frag(const unsigned short* ap) {
  v8bf lo = *(const v8bf*)(ap);
  v8bf hi = *(const v8bf*)(ap + 8);
  return __builtin_shufflevector(lo, hi, 0,1,2,3,4,5,6,7,8,9,10,11,12,13,14,15);
}

// ---------------------------------------------------------------------------
// Utility kernels
// ---------------------------------------------------------------------------
__global__ void cvt_f32_bf16_kernel(const float* __restrict__ in,
                                    unsigned short* __restrict__ out, long long n) {
  long long i = (long long)blockIdx.x * blockDim.x + threadIdx.x;
  long long s = (long long)gridDim.x * blockDim.x;
  for (; i < n; i += s) out[i] = f2bf(in[i]);
}

__global__ void mask_to_f32_kernel(const unsigned char* __restrict__ in,
                                   float* __restrict__ out, int n) {
  int i = blockIdx.x * blockDim.x + threadIdx.x;
  if (i < n) out[i] = in[i] ? 1.0f : 0.0f;
}

__global__ void init_sync_kernel(unsigned int* p) { *p = 0u; }

// ---------------------------------------------------------------------------
// Specialized bf16 GEMM:  D[m][n] = sum_k A[m][k] * W[n][KB+k]  (+ epilogue)
//   A: [M][512] bf16, W: [2048][1024] bf16, D: [M][2048] f32
//   MODE 0: + bias0[n] + bias1[n]                      (writes `base`)
//   MODE 1: + base[(m & 255)*2048 + n]                 (writes `xg`)
// Wave tile: 32(M) x 64(N): 2 shared A frags, 4 B frags, 8 accumulators.
// All strides compile-time -> loads use immediate offsets only.
// ---------------------------------------------------------------------------
template <int MODE, int KB>
__global__ __launch_bounds__(256)
void gemm_bf16_wmma_kernel(const unsigned short* __restrict__ Am,
                           const unsigned short* __restrict__ Wm,
                           int M,
                           const float* __restrict__ bias0,
                           const float* __restrict__ bias1,
                           const float* __restrict__ base,
                           float* __restrict__ D) {
  constexpr int LDA = 512;     // A row stride (elements)
  constexpr int LDB = 1024;    // W row stride (elements)
  constexpr int KK  = 512;     // reduction depth
  constexpr int NN  = 2048;    // output columns
  constexpr int LDD = 2048;    // D row stride (elements)
  constexpr int NT  = NN >> 6; // 32 n-tiles of 64

  int lane = threadIdx.x & 31;
  int wid  = (blockIdx.x * blockDim.x + threadIdx.x) >> 5;
  int total = (M >> 5) * NT;
  if (wid >= total) return;
  int mt = wid / NT;
  int nt = wid - mt * NT;
  int m0 = mt << 5;
  int n0 = nt << 6;
  int ml = lane & 15;
  int ks = lane >> 4;

  const unsigned short* ap = Am + (size_t)(m0 + ml) * LDA + 16 * ks;   // +16*LDA for half 1
  const unsigned short* bp = Wm + (size_t)(n0 + ml) * LDB + KB + 16 * ks;
  constexpr size_t ASTEP = (size_t)16 * LDA;   // 2nd m-half
  constexpr size_t BSTEP = (size_t)16 * LDB;   // 16 W rows per 16-wide N chunk

  v8f acc[8] = {v8f{}, v8f{}, v8f{}, v8f{}, v8f{}, v8f{}, v8f{}, v8f{}};
#pragma unroll 4
  for (int k = 0; k < KK; k += 32) {
    v16bf a0 = load_a_frag(ap + k);
    v16bf a1 = load_a_frag(ap + ASTEP + k);
    __builtin_prefetch(ap + k + 256, 0, 1);          // global_prefetch_b8 (A stream)
    __builtin_prefetch(ap + ASTEP + k + 256, 0, 1);
#pragma unroll
    for (int j = 0; j < 4; ++j) {
      v16bf b = *(const v16bf*)(bp + (size_t)j * BSTEP + k);
      acc[j]     = __builtin_amdgcn_wmma_f32_16x16x32_bf16(false, a0, false, b, (short)0, acc[j],     false, false);
      acc[4 + j] = __builtin_amdgcn_wmma_f32_16x16x32_bf16(false, a1, false, b, (short)0, acc[4 + j], false, false);
    }
  }

  // C/D layout: VGPR r -> m = m0 + 16*half + r + 8*ks, n = n0 + 16*j + ml
  int mbase = m0 + 8 * ks;
#pragma unroll
  for (int j = 0; j < 4; ++j) {
    int n = n0 + j * 16 + ml;
    float badd = 0.0f;
    if (MODE == 0) badd = bias0[n] + bias1[n];
#pragma unroll
    for (int half = 0; half < 2; ++half) {
#pragma unroll
      for (int r = 0; r < 8; ++r) {
        int m = mbase + 16 * half + r;
        float v = acc[4 * half + j][r];
        if (MODE == 0) v += badd;
        else           v += base[(size_t)(m & 255) * LDD + n];
        D[(size_t)m * LDD + n] = v;
      }
    }
  }
}

// ---------------------------------------------------------------------------
// Persistent recurrent kernel: all 256 LSTM steps in one launch.
// Grid: 32 blocks x 512 threads (16 waves). Block nt owns columns
// n0..n0+15 of each of the 4 gates; wave w owns batch rows 16w..16w+15.
// W_hh slice cached in LDS; c held in registers; h in global double buffer
// with a grid-wide counter barrier between steps.
// ---------------------------------------------------------------------------
__launch_bounds__(512, 1)
__global__ void lstm_persist_kernel(const unsigned short* __restrict__ Whh,
                                    const float* __restrict__ xg,
                                    unsigned short* __restrict__ h0buf,
                                    unsigned short* __restrict__ h1buf,
                                    const float* __restrict__ c0,
                                    float* __restrict__ out_hs,
                                    float* __restrict__ hN,
                                    float* __restrict__ cN,
                                    unsigned int* __restrict__ sync_ctr) {
  __shared__ __align__(32) unsigned short Wlds[64 * HH];   // 64 rows x 512 k = 64KB

  const int tid  = threadIdx.x;
  const int lane = tid & 31;
  const int wv   = tid >> 5;          // 0..15  -> m tile
  const int nt   = blockIdx.x;        // 0..31  -> j' tile
  const int n0   = nt << 4;
  const int m0   = wv << 4;
  const int ml   = lane & 15;
  const int ks   = lane >> 4;

  // --- preload W_hh rows {g*512 + n0 + i : g<4, i<16} into LDS ---
  // 4096 chunks of 8 ushorts (16B); 512 threads x 8 chunks.
  for (int ch = tid; ch < 4096; ch += 512) {
    int row = ch >> 6;                 // 64 chunks of 8 per 512-wide row
    int col = (ch & 63) * 8;
    int g = row >> 4, i = row & 15;
    *(v8bf*)(&Wlds[row * HH + col]) =
        *(const v8bf*)(Whh + (size_t)(g * HH + n0 + i) * HH + col);
  }
  __syncthreads();

  const int n     = n0 + ml;
  const int mbase = m0 + 8 * ks;

  // --- c lives in registers: lane's 8 C-layout rows, fixed column n ---
  float creg[8];
#pragma unroll
  for (int r = 0; r < 8; ++r)
    creg[r] = c0[(size_t)(mbase + r) * HH + n];

  const unsigned short* bld = &Wlds[(size_t)ml * HH + 16 * ks];   // gate g at +g*16*HH
  constexpr size_t GSTEP = (size_t)16 * HH;

  unsigned int target = 0;
  for (int t = 0; t < LSEQ; ++t) {
    const unsigned short* hin  = (t & 1) ? h1buf : h0buf;
    unsigned short*       hout = (t & 1) ? h0buf : h1buf;
    const float* xg_t = xg + (size_t)t * BB * G4;
    float*       hs_t = out_hs + (size_t)t * BB * HH;

    const unsigned short* ap = hin + (size_t)(m0 + ml) * HH + 16 * ks;

    v8f acc[4] = {v8f{}, v8f{}, v8f{}, v8f{}};
#pragma unroll 4
    for (int k = 0; k < HH; k += 32) {
      v16bf a = load_a_frag(ap + k);
#pragma unroll
      for (int g = 0; g < 4; ++g) {
        v16bf b = *(const v16bf*)(bld + (size_t)g * GSTEP + k);
        acc[g] = __builtin_amdgcn_wmma_f32_16x16x32_bf16(false, a, false, b, (short)0, acc[g], false, false);
      }
    }

#pragma unroll
    for (int r = 0; r < 8; ++r) {
      int m = mbase + r;
      const float* xr = xg_t + (size_t)m * G4 + n;
      float gi = acc[0][r] + xr[0];
      float gf = acc[1][r] + xr[HH];
      float gg = acc[2][r] + xr[2 * HH];
      float go = acc[3][r] + xr[3 * HH];
      float ii = sigmoidf_(gi);
      float ff = sigmoidf_(gf);
      float oo = sigmoidf_(go);
      float gt = tanhf(gg);
      float c1 = ff * creg[r] + ii * gt;
      creg[r] = c1;
      float h1 = oo * tanhf(c1);
      size_t idx = (size_t)m * HH + n;
      hs_t[idx]  = h1;
      hout[idx]  = f2bf(h1);
      if (t == LSEQ - 1) { hN[idx] = h1; cN[idx] = c1; }
    }

    // --- grid-wide barrier: publish h(t+1), wait for all 32 blocks ---
    __threadfence();                               // flush h writes (agent scope)
    __syncthreads();
    target += 32;                                  // gridDim.x
    if (tid == 0) {
      __hip_atomic_fetch_add(sync_ctr, 1u, __ATOMIC_RELEASE, __HIP_MEMORY_SCOPE_AGENT);
      while (__hip_atomic_load(sync_ctr, __ATOMIC_ACQUIRE, __HIP_MEMORY_SCOPE_AGENT) < target)
        __builtin_amdgcn_s_sleep(2);
    }
    __syncthreads();
    // per-wave acquire so every wave's cache path is invalidated before
    // reading h produced by other blocks
    (void)__hip_atomic_load(sync_ctr, __ATOMIC_ACQUIRE, __HIP_MEMORY_SCOPE_AGENT);
  }
}

// ---------------------------------------------------------------------------
// Workspace layout (bytes)
// ---------------------------------------------------------------------------
static constexpr size_t OFF_XG   = 0;
static constexpr size_t SZ_XG    = (size_t)LSEQ * BB * G4 * 4;       // 512 MB f32
static constexpr size_t OFF_XBF  = OFF_XG + SZ_XG;
static constexpr size_t SZ_XBF   = (size_t)LSEQ * BB * AA * 2;       // 64 MB bf16
static constexpr size_t OFF_WIH  = OFF_XBF + SZ_XBF;
static constexpr size_t SZ_WIH   = (size_t)G4 * (II + AA) * 2;       // 4 MB
static constexpr size_t OFF_WHH  = OFF_WIH + SZ_WIH;
static constexpr size_t SZ_WHH   = (size_t)G4 * HH * 2;              // 2 MB
static constexpr size_t OFF_FEAT = OFF_WHH + SZ_WHH;
static constexpr size_t SZ_FEAT  = (size_t)BB * II * 2;
static constexpr size_t OFF_BASE = OFF_FEAT + SZ_FEAT;
static constexpr size_t SZ_BASE  = (size_t)BB * G4 * 4;              // 2 MB
static constexpr size_t OFF_H0   = OFF_BASE + SZ_BASE;
static constexpr size_t SZ_H     = (size_t)BB * HH * 2;
static constexpr size_t OFF_H1   = OFF_H0 + SZ_H;
static constexpr size_t OFF_SYNC = OFF_H1 + SZ_H;

extern "C" void kernel_launch(void* const* d_in, const int* in_sizes, int n_in,
                              void* d_out, int out_size, void* d_ws, size_t ws_size,
                              hipStream_t stream) {
  (void)in_sizes; (void)n_in; (void)out_size; (void)ws_size;

  const float*         feat   = (const float*)d_in[0];          // [B, I]
  const float*         xdata  = (const float*)d_in[1];          // [L, B, A]
  const unsigned char* maskin = (const unsigned char*)d_in[2];  // [L, B] bool
  const float*         h0in   = (const float*)d_in[3];          // [B, H]
  const float*         c0in   = (const float*)d_in[4];          // [B, H]
  const float*         Wih    = (const float*)d_in[5];          // [4H, I+A]
  const float*         Whh    = (const float*)d_in[6];          // [4H, H]
  const float*         bih    = (const float*)d_in[7];          // [4H]
  const float*         bhh    = (const float*)d_in[8];          // [4H]
  float* out = (float*)d_out;

  char* ws = (char*)d_ws;
  float*          xg     = (float*)(ws + OFF_XG);
  unsigned short* Xbf    = (unsigned short*)(ws + OFF_XBF);
  unsigned short* WihBf  = (unsigned short*)(ws + OFF_WIH);
  unsigned short* WhhBf  = (unsigned short*)(ws + OFF_WHH);
  unsigned short* featBf = (unsigned short*)(ws + OFF_FEAT);
  float*          basep  = (float*)(ws + OFF_BASE);
  unsigned short* h0buf  = (unsigned short*)(ws + OFF_H0);
  unsigned short* h1buf  = (unsigned short*)(ws + OFF_H1);
  unsigned int*   syncp  = (unsigned int*)(ws + OFF_SYNC);

  // d_out layout: hs [L*B*H] | mask [L*B] | h_n [B*H] | c_n [B*H]
  float* out_hs   = out;
  float* out_mask = out + (size_t)LSEQ * BB * HH;
  float* out_hN   = out_mask + (size_t)LSEQ * BB;
  float* out_cN   = out_hN + (size_t)BB * HH;

  // --- conversions to bf16 + state init ---
  cvt_f32_bf16_kernel<<<1024, 256, 0, stream>>>(Wih,   WihBf,  (long long)G4 * (II + AA));
  cvt_f32_bf16_kernel<<<1024, 256, 0, stream>>>(Whh,   WhhBf,  (long long)G4 * HH);
  cvt_f32_bf16_kernel<<<256,  256, 0, stream>>>(feat,  featBf, (long long)BB * II);
  cvt_f32_bf16_kernel<<<2048, 256, 0, stream>>>(xdata, Xbf,    (long long)LSEQ * BB * AA);
  cvt_f32_bf16_kernel<<<256,  256, 0, stream>>>(h0in,  h0buf,  (long long)BB * HH);
  init_sync_kernel<<<1, 1, 0, stream>>>(syncp);
  mask_to_f32_kernel<<<(LSEQ * BB + 255) / 256, 256, 0, stream>>>(maskin, out_mask, LSEQ * BB);

  // --- (1) base = feat @ W_ih[:, :I]^T + b_ih + b_hh   [256 x 2048] ---
  {
    int M = BB;
    int waves = (M >> 5) * (G4 >> 6);
    gemm_bf16_wmma_kernel<0, 0><<<(waves + 7) / 8, 256, 0, stream>>>(
        featBf, WihBf, M, bih, bhh, nullptr, basep);
  }

  // --- (2) xg[t*B + b] = base[b] + x @ W_ih[:, I:]^T   [65536 x 2048] ---
  {
    int M = LSEQ * BB;
    int waves = (M >> 5) * (G4 >> 6);
    gemm_bf16_wmma_kernel<1, II><<<(waves + 7) / 8, 256, 0, stream>>>(
        Xbf, WihBf, M, nullptr, nullptr, basep, xg);
  }

  // --- (3) all 256 recurrent steps in one persistent kernel ---
  lstm_persist_kernel<<<32, 512, 0, stream>>>(
      WhhBf, xg, h0buf, h1buf, c0in, out_hs, out_hN, out_cN, syncp);
}
